// ChildSumLSTMLayer_6055903887872
// MI455X (gfx1250) — compile-verified, matrix operand order checked by reading, not run on hardware
//
#include <hip/hip_runtime.h>
#include <hip/hip_bf16.h>

#define L_DIM   8
#define N_NODES 16384
#define KCH     16
#define DIN     256
#define DOUT    256

typedef __attribute__((ext_vector_type(8)))  __bf16 v8bf;
typedef __attribute__((ext_vector_type(16))) __bf16 v16bf;
typedef __attribute__((ext_vector_type(8)))  float  v8f;
typedef __attribute__((ext_vector_type(4))) unsigned int u32x4;
typedef __attribute__((ext_vector_type(8))) int          i32x8;
typedef __attribute__((ext_vector_type(4))) int          i32x4;

// float -> bf16, round-to-nearest-even (bit-level)
static __device__ __forceinline__ __bf16 f2bf(float f) {
    unsigned u = __builtin_bit_cast(unsigned, f);
    unsigned r = (u + 0x7FFFu + ((u >> 16) & 1u)) >> 16;
    unsigned short s = (unsigned short)r;
    return __builtin_bit_cast(__bf16, s);
}

static __device__ __forceinline__ float sigmoidf_fast(float x) {
    return 1.0f / (1.0f + __expf(-x));
}

// ---------------------------------------------------------------------------
// Tensor Data Mover helpers (CDNA5 TDM, tracked by TENSORcnt).
// Descriptor layout per CDNA5 ISA.
// ---------------------------------------------------------------------------
static __device__ __forceinline__ void tdm_issue(u32x4 g0, i32x8 g1,
                                                 i32x4 g2, i32x4 g3) {
#if defined(__clang_major__) && __clang_major__ >= 23
    i32x8 z8 = {0, 0, 0, 0, 0, 0, 0, 0};
    __builtin_amdgcn_tensor_load_to_lds(g0, g1, g2, g3, z8, 0);
#else
    __builtin_amdgcn_tensor_load_to_lds(g0, g1, g2, g3, 0);
#endif
}

// 2-D bf16 tile load: tile_d1 rows x tile_d0 elements, row stride = stride0
// elements; gaddr points at the tile start. Dest = contiguous LDS.
static __device__ __forceinline__ void tdm_load_2d_bf16(
    unsigned lds_addr, const void* gaddr,
    unsigned tile_d0, unsigned tile_d1,
    unsigned tensor_d0, unsigned tensor_d1, unsigned stride0) {
    unsigned long long ga = (unsigned long long)(uintptr_t)gaddr;
    u32x4 g0;
    g0[0] = 1u;                                   // count=1, user descriptor
    g0[1] = lds_addr;                             // LDS byte address
    g0[2] = (unsigned)(ga & 0xFFFFFFFFu);         // global_addr[31:0]
    g0[3] = (unsigned)((ga >> 32) & 0x1FFFFFFu) | (2u << 30);  // type=2
    i32x8 g1;
    g1[0] = (int)(1u << 16);                      // data_size=1 (2 bytes)
    g1[1] = (int)((tensor_d0 & 0xFFFFu) << 16);   // tensor_dim0[15:0]
    g1[2] = (int)(((tensor_d0 >> 16) & 0xFFFFu) | ((tensor_d1 & 0xFFFFu) << 16));
    g1[3] = (int)(((tensor_d1 >> 16) & 0xFFFFu) | ((tile_d0 & 0xFFFFu) << 16));
    g1[4] = (int)(tile_d1 & 0xFFFFu);             // tile_dim1 (tile_dim2=0)
    g1[5] = (int)stride0;                         // tensor_dim0_stride[31:0]
    g1[6] = 0;
    g1[7] = 0;
    i32x4 z = {0, 0, 0, 0};
    tdm_issue(g0, g1, z, z);
}

// Gather-mode bf16 load: pull 16 rows (16-bit indices) of row_elems each
// from a [tensor_d1 x row_elems] tensor into contiguous LDS. Index 0xFFFF
// is out-of-bounds -> hardware returns zeros (our child mask).
static __device__ __forceinline__ void tdm_gather16_bf16(
    unsigned lds_addr, const void* gaddr, unsigned row_elems,
    unsigned tensor_d1, const unsigned short* ridx) {
    unsigned long long ga = (unsigned long long)(uintptr_t)gaddr;
    u32x4 g0;
    g0[0] = 1u | (1u << 31);                      // count=1, gather_mode=1, 16b idx
    g0[1] = lds_addr;
    g0[2] = (unsigned)(ga & 0xFFFFFFFFu);
    g0[3] = (unsigned)((ga >> 32) & 0x1FFFFFFu) | (2u << 30);
    i32x8 g1;
    g1[0] = (int)(1u << 16);                      // data_size=1 (2 bytes)
    g1[1] = (int)((row_elems & 0xFFFFu) << 16);   // tensor_dim0
    g1[2] = (int)((tensor_d1 & 0xFFFFu) << 16);   // tensor_dim1
    g1[3] = (int)((row_elems & 0xFFFFu) << 16);   // tile_dim0 = row_elems
    g1[4] = (int)(16u);                           // tile_dim1 = #indices
    g1[5] = (int)row_elems;                       // tensor_dim0_stride
    g1[6] = 0;
    g1[7] = 0;
    i32x4 g2, g3;
    #pragma unroll
    for (int w = 0; w < 4; ++w) {
        g2[w] = (int)((unsigned)ridx[2 * w] | ((unsigned)ridx[2 * w + 1] << 16));
        g3[w] = (int)((unsigned)ridx[8 + 2 * w] | ((unsigned)ridx[8 + 2 * w + 1] << 16));
    }
    tdm_issue(g0, g1, g2, g3);
}

// Load a 16x32 bf16 WMMA fragment from a 32-contiguous-halfword row slice.
// Lanes 0-15: row=lane, K 0..7 in v0..3, K 16..23 in v4..7; lanes 16-31: +8.
static __device__ __forceinline__ v16bf ldfrag(const __bf16* rowp, int lane) {
    const int base = (lane & 16) ? 8 : 0;
    v8bf lo = *(const v8bf*)(rowp + base);
    v8bf hi = *(const v8bf*)(rowp + base + 16);
    return __builtin_shufflevector(lo, hi, 0, 1, 2, 3, 4, 5, 6, 7,
                                   8, 9, 10, 11, 12, 13, 14, 15);
}

// ---------------------------------------------------------------------------
// One-shot conversion kernels
// ---------------------------------------------------------------------------
__global__ __launch_bounds__(256)
void cvt_bf16_kernel(const float* __restrict__ in, __bf16* __restrict__ out) {
    const int t = blockIdx.x * 256 + threadIdx.x;
    out[t] = f2bf(in[t]);
}

// out[n][k] = in[k][n]; K fixed at 256 (k = t&255, n = t>>8)
__global__ __launch_bounds__(256)
void transpose_cvt_kernel(const float* __restrict__ in, __bf16* __restrict__ out,
                          int Nd) {
    const int t = blockIdx.x * 256 + threadIdx.x;
    const int k = t & 255;
    const int n = t >> 8;
    out[t] = f2bf(in[(size_t)k * Nd + n]);
}

// ---------------------------------------------------------------------------
// C[M,Ndim] = A[M,K] @ Bt[Ndim,K]^T (+bias). bf16 in, f32 out.
// Block tile 64x256, 8 waves (2x4), each wave 32x64 (2x4 accumulators).
// Double-buffered TDM staging: tile k+1 streams in while tile k computes.
// ---------------------------------------------------------------------------
__global__ __launch_bounds__(256)
void gemm_bias_kernel(const __bf16* __restrict__ A, const __bf16* __restrict__ Bt,
                      const float* __restrict__ bias, float* __restrict__ C,
                      int M, int Ndim, int Kdim) {
    __shared__ __bf16 As[2][64][32];     //  8 KB
    __shared__ __bf16 Bs[2][256][32];    // 32 KB

    const int tid  = threadIdx.x;
    const int lane = tid & 31;
    const int wave = tid >> 5;
    const int wm   = wave >> 2;          // 0..1 (32-row slab)
    const int wn   = wave & 3;           // 0..3 (64-col slab)
    const int m0   = blockIdx.y * 64;
    const int n0   = blockIdx.x * 256;
    const int nk   = Kdim >> 5;

    v8f acc[2][4] = {};

    if (wave == 0) {                     // prologue: tile 0 -> buffer 0
        tdm_load_2d_bf16((unsigned)(uintptr_t)&As[0][0][0],
                         A + (size_t)m0 * Kdim, 32, 64,
                         (unsigned)Kdim, (unsigned)M, (unsigned)Kdim);
        tdm_load_2d_bf16((unsigned)(uintptr_t)&Bs[0][0][0],
                         Bt + (size_t)n0 * Kdim, 32, 256,
                         (unsigned)Kdim, (unsigned)Ndim, (unsigned)Kdim);
    }

    for (int kt = 0; kt < nk; ++kt) {
        const int p = kt & 1;
        __syncthreads();                 // prior reads of buf[1-p] done
        if (wave == 0) {
            if (kt + 1 < nk) {           // stream next tile into buf[1-p]
                const int k0n = (kt + 1) << 5;
                tdm_load_2d_bf16((unsigned)(uintptr_t)&As[1 - p][0][0],
                                 A + (size_t)m0 * Kdim + k0n, 32, 64,
                                 (unsigned)Kdim, (unsigned)M, (unsigned)Kdim);
                tdm_load_2d_bf16((unsigned)(uintptr_t)&Bs[1 - p][0][0],
                                 Bt + (size_t)n0 * Kdim + k0n, 32, 256,
                                 (unsigned)Kdim, (unsigned)Ndim, (unsigned)Kdim);
                __builtin_amdgcn_s_wait_tensorcnt(2);  // tile kt complete
            } else {
                __builtin_amdgcn_s_wait_tensorcnt(0);
            }
        }
        __syncthreads();                 // tile kt visible to all waves

        v16bf af[2];
        #pragma unroll
        for (int mi = 0; mi < 2; ++mi)
            af[mi] = ldfrag(&As[p][wm * 32 + mi * 16 + (lane & 15)][0], lane);
        #pragma unroll
        for (int ni = 0; ni < 4; ++ni) {
            v16bf bfr = ldfrag(&Bs[p][wn * 64 + ni * 16 + (lane & 15)][0], lane);
            #pragma unroll
            for (int mi = 0; mi < 2; ++mi)
                acc[mi][ni] = __builtin_amdgcn_wmma_f32_16x16x32_bf16(
                    false, af[mi], false, bfr, (short)0, acc[mi][ni], false, false);
        }
    }

    #pragma unroll
    for (int mi = 0; mi < 2; ++mi) {
        #pragma unroll
        for (int ni = 0; ni < 4; ++ni) {
            const int col   = n0 + wn * 64 + ni * 16 + (lane & 15);
            const float bv  = bias ? bias[col] : 0.0f;
            const int rbase = m0 + wm * 32 + mi * 16 + ((lane >> 4) << 3);
            #pragma unroll
            for (int r = 0; r < 8; ++r)
                C[(size_t)(rbase + r) * Ndim + col] = acc[mi][ni][r] + bv;
        }
    }
}

// ---------------------------------------------------------------------------
// Fused child kernel. One node per wave, 4 waves/block.
// Per-wave TDM gather (16 rows, masked via OOB index) -> LDS bf16 A (16x256);
// h_sum from the staged tile; F = sigma(A @ U_f + Wf); f = sum_k F_k * c_k.
// U_f panels double-buffered through TDM.
// ---------------------------------------------------------------------------
__global__ __launch_bounds__(128)
void childsum_fk_kernel(const int* __restrict__ idx_lvl,    // [N, 16]
                        const __bf16* __restrict__ hbf,     // [N, 256] prev h bf16
                        const float* __restrict__ c_prev,   // [N, 256] prev c f32
                        const __bf16* __restrict__ Uf_t,    // [256, 256] (n,k)
                        const float* __restrict__ Wx,       // [N, 1024] (Wf at 0)
                        __bf16* __restrict__ hsum_bf,       // [N, 256]
                        float* __restrict__ f_out,          // [N, 256]
                        int have_prev) {
    __shared__ __bf16 Ash[4][KCH][DIN];   // 32 KB: per-wave child matrices
    __shared__ __bf16 Bp[2][DOUT][32];    // 32 KB: double-buffered U_f panels

    const int tid  = threadIdx.x;
    const int lane = tid & 31;
    const int wave = tid >> 5;
    const int node = blockIdx.x * 4 + wave;

    if (!have_prev) {                     // level 0: all children are zero rows
        const __bf16 zb = f2bf(0.0f);
        #pragma unroll
        for (int j = 0; j < 8; ++j) {
            hsum_bf[(size_t)node * DOUT + lane + 32 * j] = zb;
            f_out[(size_t)node * DOUT + lane + 32 * j] = 0.0f;
        }
        return;
    }

    // Indices (wave-uniform scalar loads) + gather row ids (0xFFFF == masked)
    const int* idxg = idx_lvl + (size_t)node * KCH;
    int ids[KCH];
    unsigned short ridx[KCH];
    #pragma unroll
    for (int k = 0; k < KCH; ++k) {
        ids[k]  = idxg[k];
        ridx[k] = (ids[k] >= 1) ? (unsigned short)(ids[k] - 1) : (unsigned short)0xFFFFu;
    }

    // One TDM gather per wave: 16 children rows -> Ash[wave]
    tdm_gather16_bf16((unsigned)(uintptr_t)&Ash[wave][0][0], hbf,
                      DIN, N_NODES, ridx);
    __builtin_amdgcn_s_wait_tensorcnt(0);

    // Child-sum from the staged tile
    float hs[8] = {};
    for (int k = 0; k < KCH; ++k) {
        #pragma unroll
        for (int j = 0; j < 8; ++j)
            hs[j] += (float)Ash[wave][k][lane + 32 * j];
    }
    #pragma unroll
    for (int j = 0; j < 8; ++j)
        hsum_bf[(size_t)node * DOUT + lane + 32 * j] = f2bf(hs[j]);

    if (wave == 0) {                      // prologue: panel 0 -> buffer 0
        tdm_load_2d_bf16((unsigned)(uintptr_t)&Bp[0][0][0],
                         Uf_t, 32, DOUT, DIN, DOUT, DIN);
    }

    v8f acc[16] = {};
    for (int ks = 0; ks < 8; ++ks) {
        const int p = ks & 1;
        __syncthreads();                  // prior reads of Bp[1-p] done
        if (wave == 0) {
            if (ks + 1 < 8) {
                tdm_load_2d_bf16((unsigned)(uintptr_t)&Bp[1 - p][0][0],
                                 Uf_t + (ks + 1) * 32, 32, DOUT, DIN, DOUT, DIN);
                __builtin_amdgcn_s_wait_tensorcnt(1);  // panel ks complete
            } else {
                __builtin_amdgcn_s_wait_tensorcnt(0);
            }
        }
        __syncthreads();

        v16bf af = ldfrag(&Ash[wave][lane & 15][ks * 32], lane);
        #pragma unroll
        for (int nt = 0; nt < 16; ++nt) {
            v16bf bfr = ldfrag(&Bp[p][nt * 16 + (lane & 15)][0], lane);
            acc[nt] = __builtin_amdgcn_wmma_f32_16x16x32_bf16(
                false, af, false, bfr, (short)0, acc[nt], false, false);
        }
    }

    // Epilogue: sigmoid + masked c gather + child reduction.
    const float* wxrow = Wx + (size_t)node * 1024;  // Wf slice
    const int hi   = lane >> 4;
    const int colb = lane & 15;
    #pragma unroll
    for (int nt = 0; nt < 16; ++nt) {
        const int col = nt * 16 + colb;
        const float wf = wxrow[col];
        float partial = 0.0f;
        #pragma unroll
        for (int r = 0; r < 8; ++r) {
            const int id = ids[r + 8 * hi];
            const float cv = (id >= 1)
                                 ? c_prev[(size_t)(id - 1) * DOUT + col] : 0.0f;
            partial += sigmoidf_fast(acc[nt][r] + wf) * cv;
        }
        partial += __shfl_xor(partial, 16, 32);
        if (hi == 0) f_out[(size_t)node * DOUT + col] = partial;
    }
}

// ---------------------------------------------------------------------------
// Gates: i=sig(iuo0+Wi), u=tanh(iuo1+Wu), o=sig(iuo2+Wo), c=i*u+f,
// h=o*tanh(c). Writes f32 h,c to d_out and bf16 h for next level's gather.
// ---------------------------------------------------------------------------
__global__ __launch_bounds__(256)
void gates_kernel(const float* __restrict__ Wx, const float* __restrict__ iuo,
                  const float* __restrict__ fbuf, float* __restrict__ h_out,
                  float* __restrict__ c_out, __bf16* __restrict__ hbf_out) {
    const int t    = blockIdx.x * blockDim.x + threadIdx.x;
    const int node = t >> 8;
    const int col  = t & 255;
    const float* wxr = Wx  + (size_t)node * 1024;
    const float* ir  = iuo + (size_t)node * 768;
    const float i = sigmoidf_fast(ir[col]       + wxr[256 + col]);
    const float u = tanhf(       ir[256 + col]  + wxr[512 + col]);
    const float o = sigmoidf_fast(ir[512 + col] + wxr[768 + col]);
    const float c = i * u + fbuf[t];
    const float h = o * tanhf(c);
    h_out[t] = h;
    c_out[t] = c;
    hbf_out[t] = f2bf(h);
}

extern "C" void kernel_launch(void* const* d_in, const int* in_sizes, int n_in,
                              void* d_out, int out_size, void* d_ws, size_t ws_size,
                              hipStream_t stream) {
    const float* tensor  = (const float*)d_in[0];   // [L,N,256]
    const int*   indices = (const int*)  d_in[1];   // [L,N,16]
    const float* W_w     = (const float*)d_in[2];   // [256,1024]
    const float* W_b     = (const float*)d_in[3];   // [1024]
    const float* U_f     = (const float*)d_in[4];   // [256,256]
    const float* U_iuo   = (const float*)d_in[5];   // [256,768]

    float* out   = (float*)d_out;
    float* h_all = out;
    float* c_all = out + (size_t)L_DIM * N_NODES * DOUT;

    // Workspace: f32 region then bf16 region
    float* Wx   = (float*)d_ws;                        // N*1024 f32
    float* fbuf = Wx   + (size_t)N_NODES * 1024;       // N*256  f32
    float* iuo  = fbuf + (size_t)N_NODES * DOUT;       // N*768  f32
    __bf16* bfr = (__bf16*)(iuo + (size_t)N_NODES * 768);
    __bf16* Xbf     = bfr;                             // L*N*256
    __bf16* hsum_bf = Xbf     + (size_t)L_DIM * N_NODES * DIN;  // N*256
    __bf16* hbf     = hsum_bf + (size_t)N_NODES * DOUT;         // N*256
    __bf16* Wt_w    = hbf     + (size_t)N_NODES * DOUT;         // 1024*256
    __bf16* Uf_t    = Wt_w    + (size_t)1024 * DIN;             // 256*256
    __bf16* Uiuo_t  = Uf_t    + (size_t)DOUT * DIN;             // 768*256

    // One-shot conversions: activations -> bf16, weights -> transposed bf16
    cvt_bf16_kernel<<<(L_DIM * N_NODES * DIN) / 256, 256, 0, stream>>>(tensor, Xbf);
    transpose_cvt_kernel<<<(1024 * 256) / 256, 256, 0, stream>>>(W_w, Wt_w, 1024);
    transpose_cvt_kernel<<<(256 * 256) / 256, 256, 0, stream>>>(U_f, Uf_t, 256);
    transpose_cvt_kernel<<<(768 * 256) / 256, 256, 0, stream>>>(U_iuo, Uiuo_t, 768);

    for (int lev = 0; lev < L_DIM; ++lev) {
        // Wx = x @ W_w + b          [16384,256]@[256,1024]
        gemm_bias_kernel<<<dim3(1024 / 256, N_NODES / 64), 256, 0, stream>>>(
            Xbf + (size_t)lev * N_NODES * DIN, Wt_w, W_b, Wx, N_NODES, 1024, DIN);

        const float* cp = lev ? (c_all + (size_t)(lev - 1) * N_NODES * DOUT) : nullptr;

        // TDM-gathered children + h_sum + f
        childsum_fk_kernel<<<N_NODES / 4, 128, 0, stream>>>(
            indices + (size_t)lev * N_NODES * KCH, hbf, cp, Uf_t, Wx,
            hsum_bf, fbuf, lev > 0 ? 1 : 0);

        // iuo = h_sum @ U_iuo       [16384,256]@[256,768]
        gemm_bias_kernel<<<dim3(768 / 256, N_NODES / 64), 256, 0, stream>>>(
            hsum_bf, Uiuo_t, nullptr, iuo, N_NODES, 768, DIN);

        // gates + write h,c (and bf16 h for next level's TDM gather)
        gates_kernel<<<(N_NODES * DOUT) / 256, 256, 0, stream>>>(
            Wx, iuo, fbuf,
            h_all + (size_t)lev * N_NODES * DOUT,
            c_all + (size_t)lev * N_NODES * DOUT, hbf);
    }
}
